// Method_Classification_163208757261
// MI455X (gfx1250) — matrix operand order repaired and checked by reading, not compiled, not run on hardware
//
#include <hip/hip_runtime.h>
#include <hip/hip_bf16.h>

typedef __bf16 v16bf __attribute__((ext_vector_type(16)));
typedef float  v8f   __attribute__((ext_vector_type(8)));

#define GN_NODES 100000
#define GIN_DIM  1433
#define GHID     50
#define GOUT_DIM 7
#define KSTEPS   45      // ceil(1433/32); 44 full steps + 25-wide tail
#define NTILES   4       // 4*16 = 64 >= 50 output cols

// ---------------------------------------------------------------------------
// degree / normalization
// ---------------------------------------------------------------------------
__global__ void k_deg_init(float* __restrict__ deg, int n) {
    int i = blockIdx.x * blockDim.x + threadIdx.x;
    if (i < n) deg[i] = 1.0f;               // self-loop contributes 1
}

__global__ void k_deg_edges(const int* __restrict__ col, int E, float* __restrict__ deg) {
    int e = blockIdx.x * blockDim.x + threadIdx.x;
    if (e < E) atomicAdd(&deg[col[e]], 1.0f);
}

__global__ void k_make_dinv(float* __restrict__ deg, int n) {
    int i = blockIdx.x * blockDim.x + threadIdx.x;
    if (i < n) {
        float d = deg[i];
        deg[i] = (d > 0.0f) ? rsqrtf(d) : 0.0f;   // in-place: deg -> dinv
    }
}

// ---------------------------------------------------------------------------
// pack W1 [1433,50] f32 -> bf16 B-fragments in WMMA 32x16 layout
// fragment index: ((t*NTILES + nt)*32 + lane)*16 + j
// lane<16 : N = nt*16+lane,     K = t*32 + (j&7) + 16*(j>>3)
// lane>=16: N = nt*16+lane-16,  K = t*32 + 8 + (j&7) + 16*(j>>3)
// ---------------------------------------------------------------------------
__global__ void k_pack_w1(const float* __restrict__ W1, __bf16* __restrict__ pk) {
    int idx = blockIdx.x * blockDim.x + threadIdx.x;
    if (idx >= KSTEPS * NTILES * 32 * 16) return;
    int j    = idx & 15;
    int lane = (idx >> 4) & 31;
    int nt   = (idx >> 9) & 3;
    int t    = idx >> 11;
    int colN = nt * 16 + (lane & 15);
    int k    = t * 32 + ((lane >= 16) ? 8 : 0) + (j & 7) + ((j & 8) << 1);
    float v  = (k < GIN_DIM && colN < GHID) ? W1[k * GHID + colN] : 0.0f;
    pk[idx] = (__bf16)v;
}

// ---------------------------------------------------------------------------
// GEMM1: H[100000,50] = X[100000,1433] @ W1   via v_wmma_f32_16x16x32_bf16
// 1 wave -> 16 rows x 64 cols (4 accumulators). 8 waves / block.
// A fragment (16-bit A 16x32, ISA layout):
//   lanes 0-15 : M=lane,    halfs 0-7 = K 0-7,   halfs 8-15 = K 16-23
//   lanes16-31 : M=lane-16, halfs 0-7 = K 8-15,  halfs 8-15 = K 24-31
// K loop is peeled: 44 unguarded iterations + 1 guarded tail (K=1408..1432).
// ---------------------------------------------------------------------------
__global__ void __launch_bounds__(256) k_gemm1_wmma(const float* __restrict__ X,
                                                    const __bf16* __restrict__ pk,
                                                    float* __restrict__ H) {
    int wave = threadIdx.x >> 5;
    int lane = threadIdx.x & 31;
    int rowtile = blockIdx.x * 8 + wave;
    if (rowtile >= (GN_NODES / 16)) return;          // wave-uniform exit

    int m0   = lane & 15;
    int koff = (lane >> 4) << 3;                     // 0 or 8
    const float* Xr = X + (long)(rowtile * 16 + m0) * GIN_DIM;

    v8f acc0 = {}, acc1 = {}, acc2 = {}, acc3 = {};
    const v16bf* Bfrag = (const v16bf*)pk;

    // ---- 44 full, guard-free K steps ----
    for (int t = 0; t < KSTEPS - 1; ++t) {
        int kk = t * 32;
        __builtin_prefetch(Xr + kk + 64, 0, 1);      // global_prefetch_b8

        const float* p0 = Xr + kk + koff;            // K kk+koff .. +7   (contiguous)
        const float* p1 = Xr + kk + 16 + koff;       // K kk+16+koff .. +7
        v16bf a;
#pragma unroll
        for (int j = 0; j < 8; ++j) {
            a[j]     = (__bf16)p0[j];
            a[8 + j] = (__bf16)p1[j];
        }

        const v16bf* B = Bfrag + (size_t)(t * NTILES) * 32 + lane;
        v16bf b0 = B[0 * 32];
        v16bf b1 = B[1 * 32];
        v16bf b2 = B[2 * 32];
        v16bf b3 = B[3 * 32];

        acc0 = __builtin_amdgcn_wmma_f32_16x16x32_bf16(false, a, false, b0, (short)0, acc0, false, false);
        acc1 = __builtin_amdgcn_wmma_f32_16x16x32_bf16(false, a, false, b1, (short)0, acc1, false, false);
        acc2 = __builtin_amdgcn_wmma_f32_16x16x32_bf16(false, a, false, b2, (short)0, acc2, false, false);
        acc3 = __builtin_amdgcn_wmma_f32_16x16x32_bf16(false, a, false, b3, (short)0, acc3, false, false);
    }

    // ---- tail step t = 44, kk = 1408: only K >= 1433 must be zeroed ----
    {
        const int t  = KSTEPS - 1;
        const int kk = t * 32;                       // 1408
        const float* p0 = Xr + kk + koff;
        v16bf a;
#pragma unroll
        for (int j = 0; j < 8; ++j)                  // K 1408..1423: always valid
            a[j] = (__bf16)p0[j];
#pragma unroll
        for (int j = 0; j < 8; ++j) {                // K 1424..1439: guard
            int k1 = kk + 16 + koff + j;
            a[8 + j] = (k1 < GIN_DIM) ? (__bf16)Xr[k1] : (__bf16)0.0f;
        }

        const v16bf* B = Bfrag + (size_t)(t * NTILES) * 32 + lane;
        v16bf b0 = B[0 * 32];
        v16bf b1 = B[1 * 32];
        v16bf b2 = B[2 * 32];
        v16bf b3 = B[3 * 32];

        acc0 = __builtin_amdgcn_wmma_f32_16x16x32_bf16(false, a, false, b0, (short)0, acc0, false, false);
        acc1 = __builtin_amdgcn_wmma_f32_16x16x32_bf16(false, a, false, b1, (short)0, acc1, false, false);
        acc2 = __builtin_amdgcn_wmma_f32_16x16x32_bf16(false, a, false, b2, (short)0, acc2, false, false);
        acc3 = __builtin_amdgcn_wmma_f32_16x16x32_bf16(false, a, false, b3, (short)0, acc3, false, false);
    }

    // C/D layout: lanes 0-15 -> C[M=v][N=lane]; lanes 16-31 -> C[M=8+v][N=lane-16]
    int mbase = (lane >> 4) << 3;                    // 0 or 8
    int n = lane & 15;
#pragma unroll
    for (int v = 0; v < 8; ++v) {
        long grow = (long)(rowtile * 16 + mbase + v) * GHID;
        H[grow + n] = acc0[v];                                    // cols 0-15
        H[grow + 16 + n] = acc1[v];                               // cols 16-31
        if (32 + n < GHID) H[grow + 32 + n] = acc2[v];            // cols 32-47
        if (48 + n < GHID) H[grow + 48 + n] = acc3[v];            // cols 48-49
    }
}

// ---------------------------------------------------------------------------
// self-loop: dst[i,f] = src[i,f] * dinv[i]^2     (also initializes dst)
// ---------------------------------------------------------------------------
__global__ void k_selfloop(const float* __restrict__ src, const float* __restrict__ dinv,
                           float* __restrict__ dst, int feat, long total) {
    long idx = (long)blockIdx.x * blockDim.x + threadIdx.x;
    if (idx >= total) return;
    long i = idx / feat;
    float w = dinv[i];
    dst[idx] = src[idx] * w * w;
}

// ---------------------------------------------------------------------------
// per-edge scatter: dst[col] += src[row] * dinv[row]*dinv[col]
// ---------------------------------------------------------------------------
__global__ void k_edge_agg(const int* __restrict__ row, const int* __restrict__ col,
                           const float* __restrict__ dinv, const float* __restrict__ src,
                           float* __restrict__ dst, int E, int feat) {
    int e = blockIdx.x * blockDim.x + threadIdx.x;
    if (e >= E) return;
    int r = row[e], c = col[e];
    float w = dinv[r] * dinv[c];
    const float* s = src + (long)r * feat;
    float* d = dst + (long)c * feat;
    for (int f = 0; f < feat; ++f)
        atomicAdd(&d[f], s[f] * w);                  // global_atomic_add_f32 (no return)
}

// ---------------------------------------------------------------------------
// elementwise epilogues
// ---------------------------------------------------------------------------
__global__ void k_bias_relu(const float* __restrict__ agg, const float* __restrict__ b,
                            float* __restrict__ out, int feat, long total) {
    long idx = (long)blockIdx.x * blockDim.x + threadIdx.x;
    if (idx >= total) return;
    int f = (int)(idx % feat);
    out[idx] = fmaxf(agg[idx] + b[f], 0.0f);
}

__global__ void k_bias_sigmoid(const float* __restrict__ agg, const float* __restrict__ b,
                               float* __restrict__ out, int feat, long total) {
    long idx = (long)blockIdx.x * blockDim.x + threadIdx.x;
    if (idx >= total) return;
    int f = (int)(idx % feat);
    float v = agg[idx] + b[f];
    out[idx] = 1.0f / (1.0f + __expf(-v));
}

// ---------------------------------------------------------------------------
// GEMM2 (tiny): G[i,f] = sum_k H[i,k] * W2[k,f]
// ---------------------------------------------------------------------------
__global__ void k_gemm2(const float* __restrict__ H, const float* __restrict__ W2,
                        float* __restrict__ G, long total) {
    long idx = (long)blockIdx.x * blockDim.x + threadIdx.x;
    if (idx >= total) return;
    long i = idx / GOUT_DIM;
    int  f = (int)(idx % GOUT_DIM);
    const float* h = H + i * GHID;
    float s = 0.0f;
#pragma unroll
    for (int k = 0; k < GHID; ++k)
        s = fmaf(h[k], W2[k * GOUT_DIM + f], s);
    G[idx] = s;
}

// ---------------------------------------------------------------------------
extern "C" void kernel_launch(void* const* d_in, const int* in_sizes, int n_in,
                              void* d_out, int out_size, void* d_ws, size_t ws_size,
                              hipStream_t stream) {
    const float* x    = (const float*)d_in[0];
    const int*   eidx = (const int*)d_in[1];
    const float* W1   = (const float*)d_in[2];
    const float* b1   = (const float*)d_in[3];
    const float* W2   = (const float*)d_in[4];
    const float* b2   = (const float*)d_in[5];
    float* out = (float*)d_out;

    const int N = GN_NODES;
    const int E = in_sizes[1] / 2;
    const int* row = eidx;       // src
    const int* col = eidx + E;   // dst

    // workspace layout (floats)
    float*  ws   = (float*)d_ws;
    float*  dinv = ws;                                   // 100352 (deg -> dinv in-place)
    __bf16* pk   = (__bf16*)(ws + 100352);               // 92160 bf16 (46080 floats)
    float*  h1   = ws + 100352 + 46080;                  // N*50
    float*  agg1 = h1 + (long)N * GHID;                  // N*50
    float*  g2   = agg1 + (long)N * GHID;                // N*7
    float*  agg2 = g2 + (long)N * GOUT_DIM;              // N*7

    const int B = 256;
    long nh  = (long)N * GHID;       // 5,000,000
    long no  = (long)N * GOUT_DIM;   // 700,000

    // normalization
    k_deg_init <<<(N + B - 1) / B, B, 0, stream>>>(dinv, N);
    k_deg_edges<<<(E + B - 1) / B, B, 0, stream>>>(col, E, dinv);
    k_make_dinv<<<(N + B - 1) / B, B, 0, stream>>>(dinv, N);

    // layer 1: WMMA GEMM + gather/scatter aggregation + bias/relu
    k_pack_w1<<<(KSTEPS * NTILES * 32 * 16 + B - 1) / B, B, 0, stream>>>(W1, pk);
    k_gemm1_wmma<<<(N / 16 + 7) / 8, 256, 0, stream>>>(x, pk, h1);
    k_selfloop<<<(nh + B - 1) / B, B, 0, stream>>>(h1, dinv, agg1, GHID, nh);
    k_edge_agg<<<(E + B - 1) / B, B, 0, stream>>>(row, col, dinv, h1, agg1, E, GHID);
    k_bias_relu<<<(nh + B - 1) / B, B, 0, stream>>>(agg1, b1, h1, GHID, nh);  // h1 <- relu(agg1+b1)

    // layer 2: small GEMM + aggregation + bias/sigmoid
    k_gemm2<<<(no + B - 1) / B, B, 0, stream>>>(h1, W2, g2, no);
    k_selfloop<<<(no + B - 1) / B, B, 0, stream>>>(g2, dinv, agg2, GOUT_DIM, no);
    k_edge_agg<<<(E + B - 1) / B, B, 0, stream>>>(row, col, dinv, g2, agg2, E, GOUT_DIM);
    k_bias_sigmoid<<<(no + B - 1) / B, B, 0, stream>>>(agg2, b2, out, GOUT_DIM, no);
}
